// MultiHeadAttention_36060545417219
// MI455X (gfx1250) — compile-verified, matrix operand order checked by reading, not run on hardware
//
#include <hip/hip_runtime.h>

typedef __attribute__((ext_vector_type(16))) __bf16 v16bf;
typedef __attribute__((ext_vector_type(8)))  __bf16 v8bf;
typedef __attribute__((ext_vector_type(8)))  float  v8f;

// ---------------------------------------------------------------------------
// Fragment loader: per CDNA5 ISA §7.12.2, a 16-bit A/B fragment holds, per
// lane, two 8-element (16-byte) contiguous chunks along the contraction dim:
// lanes 0-15 get K = {0..7, 16..23}, lanes 16-31 get K = {8..15, 24..31}.
// Caller passes the lane's pointer to its first chunk; second chunk is +16.
// ---------------------------------------------------------------------------
__device__ __forceinline__ v16bf load_frag16(const __bf16* p) {
    v8bf lo = *(const v8bf*)p;
    v8bf hi = *(const v8bf*)(p + 16);
    v16bf r;
#pragma unroll
    for (int i = 0; i < 8; ++i) { r[i] = lo[i]; r[i + 8] = hi[i]; }
    return r;
}

__device__ __forceinline__ v8f wmma_bf16(v16bf a, v16bf b, v8f c) {
    // D = A(16x32 bf16) * B(32x16 bf16) + C(16x16 f32)
    return __builtin_amdgcn_wmma_f32_16x16x32_bf16(
        /*neg_a=*/false, a, /*neg_b=*/false, b,
        /*c_mod=*/(short)0, c, /*reuse_a=*/false, /*reuse_b=*/false);
}

// ---------------------------------------------------------------------------
// CDNA5 async global->LDS copy (ASYNCcnt path). 16 bytes per lane.
// lds_off is the wave-relative LDS byte address (low 32 bits of a generic
// pointer into __shared__), gaddr the per-lane global source address.
// ---------------------------------------------------------------------------
__device__ __forceinline__ void async_copy_b128(unsigned lds_off, const __bf16* gaddr) {
    asm volatile("global_load_async_to_lds_b128 %0, %1, off"
                 :: "v"(lds_off), "v"(gaddr) : "memory");
}
__device__ __forceinline__ void wait_async0() {
    asm volatile("s_wait_asynccnt 0x0" ::: "memory");
}

// ---------------------------------------------------------------------------
// f32 -> bf16 conversion (grid-stride)
// ---------------------------------------------------------------------------
__global__ void mha_cvt_bf16(const float* __restrict__ in,
                             __bf16* __restrict__ out, int n) {
    int i = blockIdx.x * blockDim.x + threadIdx.x;
    int stride = gridDim.x * blockDim.x;
    for (; i < n; i += stride) out[i] = (__bf16)in[i];
}

// ---------------------------------------------------------------------------
// Double-angle RoPE cache: applying RoPE twice == rotate by 2*theta.
// cos2/sin2 layout [S][32]
// ---------------------------------------------------------------------------
__global__ void mha_rope2_cache(float* __restrict__ cos2,
                                float* __restrict__ sin2, int S) {
    int i = blockIdx.x * blockDim.x + threadIdx.x;
    if (i >= S * 32) return;
    int s = i >> 5, d = i & 31;
    float inv_freq = powf(10000.0f, -(float)d / 32.0f);
    float th = 2.0f * (float)s * inv_freq;
    cos2[i] = cosf(th);
    sin2[i] = sinf(th);
}

// ---------------------------------------------------------------------------
// Apply double-angle RoPE to [B,S,E] bf16, repack to [B,H,S,HD] bf16.
// Pairs are (d, d+32) within each 64-wide head. 'scale' folds 1/sqrt(HD)
// into Q.
// ---------------------------------------------------------------------------
__global__ void mha_rope_apply2(const __bf16* __restrict__ X,
                                const float* __restrict__ cos2,
                                const float* __restrict__ sin2,
                                __bf16* __restrict__ out,
                                float scale, int B, int S) {
    int i = blockIdx.x * blockDim.x + threadIdx.x;   // over B*S*H*32
    int total = B * S * 16 * 32;
    if (i >= total) return;
    int d = i & 31;
    int t = i >> 5;
    int h = t & 15; t >>= 4;
    int s = t % S;
    int b = t / S;
    size_t base = ((size_t)b * S + s) * 1024 + h * 64 + d;
    float x1 = (float)X[base];
    float x2 = (float)X[base + 32];
    float c  = cos2[s * 32 + d];
    float sn = sin2[s * 32 + d];
    float o1 = (x1 * c - x2 * sn) * scale;
    float o2 = (x2 * c + x1 * sn) * scale;
    size_t ob = (((size_t)(b * 16 + h)) * S + s) * 64 + d;
    out[ob]      = (__bf16)o1;
    out[ob + 32] = (__bf16)o2;
}

// ---------------------------------------------------------------------------
// V: [B,S,E] bf16 -> Vt [B,H,HD,S] bf16 so PV B-fragments are contiguous
// along the contraction (s) dimension.
// ---------------------------------------------------------------------------
__global__ void mha_v_transpose(const __bf16* __restrict__ V,
                                __bf16* __restrict__ Vt, int B, int S) {
    int i = blockIdx.x * blockDim.x + threadIdx.x;   // over B*S*E
    int total = B * S * 1024;
    if (i >= total) return;
    int e = i & 1023;
    int t = i >> 10;
    int s = t % S;
    int b = t / S;
    int h = e >> 6, d = e & 63;
    Vt[(((size_t)(b * 16 + h)) * 64 + d) * S + s] = V[i];
}

// ---------------------------------------------------------------------------
// GEMM: C[M,N] = A[M,K](bf16) * W[N,K]^T(bf16) + bias(f32)
// Block = 256 threads = 8 waves (2 M-waves x 4 N-waves); wave tile is
// 32(M) x 64(N) = 8 C-fragments, 8 WMMAs per 32-wide k-step.
// Pipelining:
//   - A: shared 64x32 tile double-buffered in LDS, staged one k-step ahead
//     with global_load_async_to_lds_b128 (ASYNCcnt) by waves 0-1.
//   - B: double-buffered in VGPRs; B(k+32) loads are issued BEFORE the WMMA
//     group for k, so the matrix ops only pay a partial s_wait_loadcnt.
// ---------------------------------------------------------------------------
__global__ __launch_bounds__(256)
void mha_gemm_wmma(const __bf16* __restrict__ A,
                   const __bf16* __restrict__ W,
                   const float* __restrict__ bias,
                   void* __restrict__ out,
                   int M, int N, int K, int out_f32) {
    __shared__ __align__(16) __bf16 Abuf[2][64][32];
    const int lane = threadIdx.x & 31;
    const int wave = threadIdx.x >> 5;
    const int wm = wave & 1, wn = wave >> 1;
    const int mblk = blockIdx.x * 64;
    const int n0 = blockIdx.y * 256 + wn * 64;
    const int half = lane >> 4;
    const int r = lane & 15;

    v8f acc[2][4] = {};
    const __bf16* wp = W + (size_t)(n0 + r) * K + half * 8;

    // staging: waves 0-1 (wave-uniform, force scalar branch), one row per
    // lane, 4 x 16B chunks per row of the 64x32 A tile
    const bool stager = (__builtin_amdgcn_readfirstlane(wave) < 2);
    const int srow = wave * 32 + lane;                 // 0..63 (waves 0-1)
    const __bf16* arow = A + (size_t)(mblk + srow) * K;

    // prologue: stage A(k=0) into buffer 0; preload B(k=0) into registers
    if (stager) {
#pragma unroll
        for (int j = 0; j < 4; ++j)
            async_copy_b128((unsigned)(uintptr_t)&Abuf[0][srow][j * 8], arow + j * 8);
    }
    v16bf b0 = load_frag16(wp + (size_t)0 * 16 * K);
    v16bf b1 = load_frag16(wp + (size_t)1 * 16 * K);
    v16bf b2 = load_frag16(wp + (size_t)2 * 16 * K);
    v16bf b3 = load_frag16(wp + (size_t)3 * 16 * K);

    int cur = 0;
    for (int k = 0; k < K; k += 32) {
        wait_async0();           // stagers: A(k) landed; others: no-op
        __syncthreads();         // publish Abuf[cur]; all done with Abuf[cur^1]
        if (stager && (k + 32 < K)) {
#pragma unroll
            for (int j = 0; j < 4; ++j)
                async_copy_b128((unsigned)(uintptr_t)&Abuf[cur ^ 1][srow][j * 8],
                                arow + (k + 32) + j * 8);
        }
        // A fragments from LDS (shared across the 4 N-waves of this M row)
        v16bf a0 = load_frag16(&Abuf[cur][wm * 32 + r][half * 8]);
        v16bf a1 = load_frag16(&Abuf[cur][wm * 32 + 16 + r][half * 8]);
        // issue next-step B loads BEFORE the WMMA group (reg double buffer)
        int kn = (k + 32 < K) ? (k + 32) : k;
        v16bf nb0 = load_frag16(wp + (size_t)0 * 16 * K + kn);
        v16bf nb1 = load_frag16(wp + (size_t)1 * 16 * K + kn);
        v16bf nb2 = load_frag16(wp + (size_t)2 * 16 * K + kn);
        v16bf nb3 = load_frag16(wp + (size_t)3 * 16 * K + kn);
        acc[0][0] = wmma_bf16(a0, b0, acc[0][0]);
        acc[1][0] = wmma_bf16(a1, b0, acc[1][0]);
        acc[0][1] = wmma_bf16(a0, b1, acc[0][1]);
        acc[1][1] = wmma_bf16(a1, b1, acc[1][1]);
        acc[0][2] = wmma_bf16(a0, b2, acc[0][2]);
        acc[1][2] = wmma_bf16(a1, b2, acc[1][2]);
        acc[0][3] = wmma_bf16(a0, b3, acc[0][3]);
        acc[1][3] = wmma_bf16(a1, b3, acc[1][3]);
        b0 = nb0; b1 = nb1; b2 = nb2; b3 = nb3;
        cur ^= 1;
    }

#pragma unroll
    for (int im = 0; im < 2; ++im) {
#pragma unroll
        for (int t = 0; t < 4; ++t) {
            int n = n0 + t * 16 + r;
            float bv = bias[n];
#pragma unroll
            for (int j = 0; j < 8; ++j) {
                int m = mblk + wm * 32 + im * 16 + j + half * 8;  // C row layout
                float v = acc[im][t][j] + bv;
                if (out_f32) ((float*)out)[(size_t)m * N + n] = v;
                else         ((__bf16*)out)[(size_t)m * N + n] = (__bf16)v;
            }
        }
    }
}

// ---------------------------------------------------------------------------
// Flash attention: grid (S/64, B*H), block 128 = 4 waves. Each wave owns a
// 16-row q tile; loops over key blocks of 32 with online softmax. Row stats
// live per-lane and match both the scores and O accumulator C-layouts
// (vgpr j <-> row j + 8*(lane>=16)). P is transposed C-layout -> A-layout
// through a per-wave 16x32 bf16 LDS buffer (LDS ops are in-order per wave).
// Q is pre-scaled by 1/sqrt(HD).
// ---------------------------------------------------------------------------
__global__ __launch_bounds__(128)
void mha_flash_attn(const __bf16* __restrict__ Qr,   // [B,H,S,HD]
                    const __bf16* __restrict__ Kr,   // [B,H,S,HD]
                    const __bf16* __restrict__ Vt,   // [B,H,HD,S]
                    __bf16* __restrict__ Out,        // [B,S,E]
                    int S) {
    __shared__ __align__(16) __bf16 Pl[4][16][32];
    const int lane = threadIdx.x & 31;
    const int w = threadIdx.x >> 5;
    const int half = lane >> 4;
    const int r = lane & 15;
    const int bh = blockIdx.y;                 // b*H + h
    const int q0 = blockIdx.x * 64 + w * 16;

    const __bf16* qbase = Qr + (size_t)bh * S * 64;
    const __bf16* kbase = Kr + (size_t)bh * S * 64;
    const __bf16* vbase = Vt + (size_t)bh * 64 * S;

    // Q tile A-fragments (16 rows x HD=64 -> two 16x32 fragments), resident.
    v16bf qa0 = load_frag16(qbase + (size_t)(q0 + r) * 64 + half * 8);
    v16bf qa1 = load_frag16(qbase + (size_t)(q0 + r) * 64 + 32 + half * 8);

    float mrow[8], lrow[8];
#pragma unroll
    for (int j = 0; j < 8; ++j) { mrow[j] = -3.0e38f; lrow[j] = 0.0f; }
    v8f o0 = {}, o1 = {}, o2 = {}, o3 = {};

    for (int kb = 0; kb < S; kb += 32) {
        // prefetch next key block rows
        __builtin_prefetch(kbase + (size_t)(kb + 32 + r) * 64, 0, 1);
        __builtin_prefetch(vbase + (size_t)r * S + kb + 32, 0, 1);
        // scores 16x32 = two C fragments; contraction over HD=64.
        // Load all four K B-fragments first, then the WMMA group.
        v8f s0 = {}, s1 = {};
        v16bf b00 = load_frag16(kbase + (size_t)(kb + r) * 64 + half * 8);
        v16bf b01 = load_frag16(kbase + (size_t)(kb + r) * 64 + 32 + half * 8);
        v16bf b10 = load_frag16(kbase + (size_t)(kb + 16 + r) * 64 + half * 8);
        v16bf b11 = load_frag16(kbase + (size_t)(kb + 16 + r) * 64 + 32 + half * 8);
        s0 = wmma_bf16(qa0, b00, s0);
        s1 = wmma_bf16(qa0, b10, s1);
        s0 = wmma_bf16(qa1, b01, s0);
        s1 = wmma_bf16(qa1, b11, s1);
        // online softmax per row (rows of C live across vgpr j + lane-half)
#pragma unroll
        for (int j = 0; j < 8; ++j) {
            float v = fmaxf(s0[j], s1[j]);
#pragma unroll
            for (int off = 1; off < 16; off <<= 1)
                v = fmaxf(v, __shfl_xor(v, off, 32));   // stays in lane-half
            float mn = fmaxf(mrow[j], v);
            float f  = __expf(mrow[j] - mn);
            float p0 = __expf(s0[j] - mn);
            float p1 = __expf(s1[j] - mn);
            float rs = p0 + p1;
#pragma unroll
            for (int off = 1; off < 16; off <<= 1)
                rs += __shfl_xor(rs, off, 32);
            lrow[j] = lrow[j] * f + rs;
            mrow[j] = mn;
            o0[j] *= f; o1[j] *= f; o2[j] *= f; o3[j] *= f;
            int row = j + half * 8;
            Pl[w][row][r]      = (__bf16)p0;   // keys kb..kb+15
            Pl[w][row][16 + r] = (__bf16)p1;   // keys kb+16..kb+31
        }
        // read P back in A-layout (same-wave LDS ops are in order), and all
        // four V B-fragments (column d, contiguous along s), then WMMA group.
        v16bf pa = load_frag16(&Pl[w][r][half * 8]);
        v16bf v0 = load_frag16(vbase + (size_t)(0 * 16 + r) * S + kb + half * 8);
        v16bf v1 = load_frag16(vbase + (size_t)(1 * 16 + r) * S + kb + half * 8);
        v16bf v2 = load_frag16(vbase + (size_t)(2 * 16 + r) * S + kb + half * 8);
        v16bf v3 = load_frag16(vbase + (size_t)(3 * 16 + r) * S + kb + half * 8);
        o0 = wmma_bf16(pa, v0, o0);
        o1 = wmma_bf16(pa, v1, o1);
        o2 = wmma_bf16(pa, v2, o2);
        o3 = wmma_bf16(pa, v3, o3);
    }

    // normalize and write [B,S,E] bf16 (input to the output projection GEMM)
    int b = bh >> 4, h = bh & 15;
    v8f oo[4] = { o0, o1, o2, o3 };
#pragma unroll
    for (int t = 0; t < 4; ++t) {
#pragma unroll
        for (int j = 0; j < 8; ++j) {
            int srow = q0 + j + half * 8;
            float val = oo[t][j] / lrow[j];
            Out[((size_t)b * S + srow) * 1024 + h * 64 + t * 16 + r] = (__bf16)val;
        }
    }
}

// ---------------------------------------------------------------------------
// Host-side orchestration
// ---------------------------------------------------------------------------
extern "C" void kernel_launch(void* const* d_in, const int* in_sizes, int n_in,
                              void* d_out, int out_size, void* d_ws, size_t ws_size,
                              hipStream_t stream) {
    (void)in_sizes; (void)n_in; (void)out_size; (void)ws_size;
    constexpr int B = 2, S = 2048, E = 1024;
    constexpr int M = B * S, N = E, K = E;

    const float* q  = (const float*)d_in[0];
    const float* k  = (const float*)d_in[1];
    const float* v  = (const float*)d_in[2];
    const float* Wq = (const float*)d_in[3];
    const float* bq = (const float*)d_in[4];
    const float* Wk = (const float*)d_in[5];
    const float* bk = (const float*)d_in[6];
    const float* Wv = (const float*)d_in[7];
    const float* bv = (const float*)d_in[8];
    const float* Wo = (const float*)d_in[9];
    const float* bo = (const float*)d_in[10];
    float* out = (float*)d_out;

    // workspace layout (all offsets 256B-aligned)
    char* ws = (char*)d_ws;
    const size_t MiB = 1024ull * 1024ull;
    __bf16* Xq  = (__bf16*)(ws + 0 * MiB);    // 8 MiB each
    __bf16* Xk  = (__bf16*)(ws + 8 * MiB);
    __bf16* Xv  = (__bf16*)(ws + 16 * MiB);
    __bf16* Wqb = (__bf16*)(ws + 24 * MiB);   // 2 MiB each
    __bf16* Wkb = (__bf16*)(ws + 26 * MiB);
    __bf16* Wvb = (__bf16*)(ws + 28 * MiB);
    __bf16* Wob = (__bf16*)(ws + 30 * MiB);
    __bf16* Qp  = (__bf16*)(ws + 32 * MiB);   // projection outputs [B,S,E]
    __bf16* Kp  = (__bf16*)(ws + 40 * MiB);
    __bf16* Vp  = (__bf16*)(ws + 48 * MiB);
    __bf16* Qr  = (__bf16*)(ws + 56 * MiB);   // [B,H,S,HD]
    __bf16* Kr  = (__bf16*)(ws + 64 * MiB);
    __bf16* Vt  = (__bf16*)(ws + 72 * MiB);   // [B,H,HD,S]
    float*  cos2 = (float*)(ws + 80 * MiB);   // 256 KiB each
    float*  sin2 = (float*)(ws + 80 * MiB + 256 * 1024);
    __bf16* attn = Xq;                        // reuse (Xq dead by then)

    // 1) convert everything to bf16
    mha_cvt_bf16<<<2048, 256, 0, stream>>>(q,  Xq,  M * E);
    mha_cvt_bf16<<<2048, 256, 0, stream>>>(k,  Xk,  M * E);
    mha_cvt_bf16<<<2048, 256, 0, stream>>>(v,  Xv,  M * E);
    mha_cvt_bf16<<<1024, 256, 0, stream>>>(Wq, Wqb, N * K);
    mha_cvt_bf16<<<1024, 256, 0, stream>>>(Wk, Wkb, N * K);
    mha_cvt_bf16<<<1024, 256, 0, stream>>>(Wv, Wvb, N * K);
    mha_cvt_bf16<<<1024, 256, 0, stream>>>(Wo, Wob, N * K);

    // 2) double-angle RoPE cache
    mha_rope2_cache<<<(S * 32 + 255) / 256, 256, 0, stream>>>(cos2, sin2, S);

    // 3) Q/K/V projections (WMMA, async-LDS A staging, reg-pipelined B)
    dim3 gg(M / 64, N / 256);
    mha_gemm_wmma<<<gg, 256, 0, stream>>>(Xq, Wqb, bq, Qp, M, N, K, 0);
    mha_gemm_wmma<<<gg, 256, 0, stream>>>(Xk, Wkb, bk, Kp, M, N, K, 0);
    mha_gemm_wmma<<<gg, 256, 0, stream>>>(Xv, Wvb, bv, Vp, M, N, K, 0);

    // 4) RoPE(x2) + repack; fold 1/sqrt(64) into Q. Transpose V.
    int nrope = B * S * 16 * 32;
    mha_rope_apply2<<<(nrope + 255) / 256, 256, 0, stream>>>(Qp, cos2, sin2, Qr, 0.125f, B, S);
    mha_rope_apply2<<<(nrope + 255) / 256, 256, 0, stream>>>(Kp, cos2, sin2, Kr, 1.0f,   B, S);
    int ntr = B * S * E;
    mha_v_transpose<<<(ntr + 255) / 256, 256, 0, stream>>>(Vp, Vt, B, S);

    // 5) flash attention (WMMA QK^T and PV, online softmax)
    mha_flash_attn<<<dim3(S / 64, B * 16), 128, 0, stream>>>(Qr, Kr, Vt, attn, S);

    // 6) output projection, f32 result
    mha_gemm_wmma<<<gg, 256, 0, stream>>>(attn, Wob, bo, (void*)out, M, N, K, 1);
}
